// sLSTMCell_37838661878325
// MI455X (gfx1250) — matrix operand order repaired and checked by reading, not compiled, not run on hardware
//
#include <hip/hip_runtime.h>
#include <hip/hip_bf16.h>

typedef __bf16 bf16_t;
typedef __attribute__((ext_vector_type(16))) __bf16 v16bf;
typedef __attribute__((ext_vector_type(8)))  __bf16 v8bf;
typedef __attribute__((ext_vector_type(4)))  __bf16 v4bf;
typedef __attribute__((ext_vector_type(8)))  float  v8f;

#define B_SZ  4096
#define IN_SZ 1024
#define H_SZ  2048
#define D_SZ  (IN_SZ + H_SZ)   // 3072
#define G_SZ  4

#define BM 64
#define BN 64
#define BK 32

__device__ __forceinline__ float sigmoidf_(float x) {
    return 1.0f / (1.0f + __expf(-x));
}

// ---------------------------------------------------------------------------
// Pack combined = concat(x, h_prev) into bf16 [B, D]
// grid: (D/1024, B), block: 256, 4 elems/thread
// ---------------------------------------------------------------------------
__global__ __launch_bounds__(256) void pack_combined(
        const float* __restrict__ x, const float* __restrict__ h_prev,
        bf16_t* __restrict__ out) {
    const int b = blockIdx.y;
    const int d = (blockIdx.x * 256 + threadIdx.x) * 4;
    float4 v;
    if (d < IN_SZ) {
        v = *(const float4*)(x + (size_t)b * IN_SZ + d);
    } else {
        v = *(const float4*)(h_prev + (size_t)b * H_SZ + (d - IN_SZ));
    }
    v4bf o = { (bf16_t)v.x, (bf16_t)v.y, (bf16_t)v.z, (bf16_t)v.w };
    *(v4bf*)(out + (size_t)b * D_SZ + d) = o;
}

// ---------------------------------------------------------------------------
// Pack Wi/Wf/Wo/Wz [H, D] fp32 -> bf16 [4, H, D]
// grid: (H*D/1024, 4), block: 256, 4 elems/thread
// ---------------------------------------------------------------------------
__global__ __launch_bounds__(256) void pack_weights(
        const float* __restrict__ Wi, const float* __restrict__ Wf,
        const float* __restrict__ Wo, const float* __restrict__ Wz,
        bf16_t* __restrict__ W4) {
    const int g = blockIdx.y;
    const float* src = (g == 0) ? Wi : (g == 1) ? Wf : (g == 2) ? Wo : Wz;
    const size_t i = ((size_t)blockIdx.x * 256 + threadIdx.x) * 4;
    float4 v = *(const float4*)(src + i);
    v4bf o = { (bf16_t)v.x, (bf16_t)v.y, (bf16_t)v.z, (bf16_t)v.w };
    *(v4bf*)(W4 + (size_t)g * H_SZ * D_SZ + i) = o;
}

// ---------------------------------------------------------------------------
// Fused 4-gate GEMM (bf16 WMMA, fp32 acc) + sLSTM state update.
// Workgroup tile: 64 rows (b) x 64 cols (h), all 4 gates.
// Double-buffered LDS + software-pipelined global fetch:
//   iter k: barrier; issue global loads for tile k+1 into regs;
//           batch-load all frags of tile k from LDS; 8 WMMAs;
//           ds_store tile k+1 into the other buffer.
// One barrier per K-step. 96 K-steps of BK=32.
// ---------------------------------------------------------------------------
__global__ __launch_bounds__(256) void slstm_gemm(
        const bf16_t* __restrict__ Acomb, const bf16_t* __restrict__ W4,
        const float* __restrict__ bi, const float* __restrict__ bfv,
        const float* __restrict__ bo, const float* __restrict__ bz,
        const float* __restrict__ c_prev, const float* __restrict__ n_prev,
        const float* __restrict__ m_prev, float* __restrict__ out) {
    __shared__ __align__(16) bf16_t sA[2][BM * BK];            // 2 x 4 KB
    __shared__ __align__(16) bf16_t sB[2][G_SZ * BN * BK];     // 2 x 16 KB

    const int t    = threadIdx.x;
    const int lane = t & 31;
    const int wave = t >> 5;
    const int wm   = wave & 3;   // 0..3 : row group of 16
    const int wn   = wave >> 2;  // 0..1 : col group of 32
    const int hw   = lane >> 4;  // half-wave select
    const int lm   = lane & 15;

    const int bBase = blockIdx.y * BM;
    const int hBase = blockIdx.x * BN;

    v8f zero8 = {0.f, 0.f, 0.f, 0.f, 0.f, 0.f, 0.f, 0.f};
    v8f acc[G_SZ][2];
    #pragma unroll
    for (int g = 0; g < G_SZ; ++g)
        #pragma unroll
        for (int nt = 0; nt < 2; ++nt)
            acc[g][nt] = zero8;

    // Staging mapping: each thread moves one 16B chunk of A and one 16B chunk
    // per gate of B each K-step.
    const int la_row = t >> 2;           // 0..63
    const int la_k   = (t & 3) * 8;      // 0,8,16,24
    const bf16_t* gA = Acomb + (size_t)(bBase + la_row) * D_SZ + la_k;
    const bf16_t* gB[G_SZ];
    #pragma unroll
    for (int g = 0; g < G_SZ; ++g)
        gB[g] = W4 + ((size_t)g * H_SZ + hBase + la_row) * D_SZ + la_k;

    // Register staging for the software pipeline
    v8bf rA;
    v8bf rB[G_SZ];

    // ---- prologue: fetch tile 0, store into buffer 0 ----
    rA = *(const v8bf*)(gA);
    #pragma unroll
    for (int g = 0; g < G_SZ; ++g) rB[g] = *(const v8bf*)(gB[g]);

    *(v8bf*)(&sA[0][la_row * BK + la_k]) = rA;
    #pragma unroll
    for (int g = 0; g < G_SZ; ++g)
        *(v8bf*)(&sB[0][(g * BN + la_row) * BK + la_k]) = rB[g];

    int buf = 0;
    for (int k0 = 0; k0 < D_SZ; k0 += BK) {
        __syncthreads();
        const bool more = (k0 + BK) < D_SZ;

        // Issue next tile's global loads early (overlap with WMMA below).
        if (more) {
            rA = *(const v8bf*)(gA + k0 + BK);
            #pragma unroll
            for (int g = 0; g < G_SZ; ++g)
                rB[g] = *(const v8bf*)(gB[g] + k0 + BK);
        }

        const bf16_t* pA = &sA[buf][0];
        const bf16_t* pB = &sB[buf][0];

        // A fragment (ISA 16-bit A 16x32 layout):
        // lane (hw,lm): elems 0..7 = K[hw*8 .. +7], elems 8..15 = K[16+hw*8 .. +7]
        union { v16bf v; v8bf h[2]; } af;
        const int arow = wm * 16 + lm;
        af.h[0] = *(const v8bf*)(pA + arow * BK + hw * 8);
        af.h[1] = *(const v8bf*)(pA + arow * BK + 16 + hw * 8);

        // Batch-load ALL B fragments first (distinct registers) so the DS
        // loads pipeline instead of serializing against each WMMA.
        union { v16bf v; v8bf h[2]; } bfr[G_SZ][2];
        #pragma unroll
        for (int g = 0; g < G_SZ; ++g) {
            #pragma unroll
            for (int nt = 0; nt < 2; ++nt) {
                const int n = wn * 32 + nt * 16 + lm;
                const bf16_t* p = pB + (g * BN + n) * BK + hw * 8;
                bfr[g][nt].h[0] = *(const v8bf*)(p);
                bfr[g][nt].h[1] = *(const v8bf*)(p + 16);
            }
        }

        #pragma unroll
        for (int g = 0; g < G_SZ; ++g)
            #pragma unroll
            for (int nt = 0; nt < 2; ++nt)
                acc[g][nt] = __builtin_amdgcn_wmma_f32_16x16x32_bf16(
                    false, af.v, false, bfr[g][nt].v, (short)0, acc[g][nt],
                    false, false);

        // Store next tile into the other buffer (global loads have had the
        // whole WMMA block to complete).
        if (more) {
            const int nb = buf ^ 1;
            *(v8bf*)(&sA[nb][la_row * BK + la_k]) = rA;
            #pragma unroll
            for (int g = 0; g < G_SZ; ++g)
                *(v8bf*)(&sB[nb][(g * BN + la_row) * BK + la_k]) = rB[g];
        }
        buf ^= 1;
    }

    // Epilogue: fused gate nonlinearities + state update.
    // C/D layout: VGPR r, lane (hw,lm) -> M = r + 8*hw, N = lm
    const size_t BH = (size_t)B_SZ * H_SZ;
    #pragma unroll
    for (int nt = 0; nt < 2; ++nt) {
        const int h = hBase + wn * 32 + nt * 16 + lm;
        const float bbi = bi[h], bbf = bfv[h], bbo = bo[h], bbz = bz[h];
        #pragma unroll
        for (int r = 0; r < 8; ++r) {
            const int brow = bBase + wm * 16 + hw * 8 + r;
            const size_t idx = (size_t)brow * H_SZ + h;
            const float gi = acc[0][nt][r] + bbi;
            const float gf = acc[1][nt][r] + bbf;
            const float go = acc[2][nt][r] + bbo;
            const float gz = acc[3][nt][r] + bbz;
            const float i_t = __expf(gi);
            const float f_t = sigmoidf_(gf);
            const float o_t = sigmoidf_(go);
            const float z_t = tanhf(gz);
            const float c_t = f_t * c_prev[idx] + i_t * z_t;
            const float n_t = f_t * n_prev[idx] + i_t;
            const float m_t = fmaxf(f_t * m_prev[idx], fabsf(i_t * z_t));
            const float h_pre = o_t * (c_t / (n_t + 1e-6f));
            out[idx]          = h_pre;   // pre-LN h (overwritten by LN kernel)
            out[BH + idx]     = c_t;
            out[2 * BH + idx] = n_t;
            out[3 * BH + idx] = m_t;
        }
    }
}

// ---------------------------------------------------------------------------
// In-place LayerNorm over last dim (H=2048). One block per row, 8 elems/thread.
// ---------------------------------------------------------------------------
__global__ __launch_bounds__(256) void layernorm_kernel(
        float* __restrict__ h, const float* __restrict__ gamma,
        const float* __restrict__ beta) {
    __shared__ float s1[256];
    __shared__ float s2[256];
    const int b = blockIdx.x;
    const int t = threadIdx.x;
    float* row = h + (size_t)b * H_SZ;

    float v[8];
    float sum = 0.f, sq = 0.f;
    #pragma unroll
    for (int j = 0; j < 8; ++j) {
        v[j] = row[t + j * 256];
        sum += v[j];
        sq  += v[j] * v[j];
    }
    s1[t] = sum;
    s2[t] = sq;
    __syncthreads();
    for (int off = 128; off > 0; off >>= 1) {
        if (t < off) { s1[t] += s1[t + off]; s2[t] += s2[t + off]; }
        __syncthreads();
    }
    const float mu  = s1[0] * (1.0f / H_SZ);
    const float var = s2[0] * (1.0f / H_SZ) - mu * mu;
    const float rs  = rsqrtf(var + 1e-5f);
    #pragma unroll
    for (int j = 0; j < 8; ++j) {
        const int hc = t + j * 256;
        row[hc] = (v[j] - mu) * rs * gamma[hc] + beta[hc];
    }
}

// ---------------------------------------------------------------------------
extern "C" void kernel_launch(void* const* d_in, const int* in_sizes, int n_in,
                              void* d_out, int out_size, void* d_ws, size_t ws_size,
                              hipStream_t stream) {
    const float* x      = (const float*)d_in[0];
    const float* h_prev = (const float*)d_in[1];
    const float* c_prev = (const float*)d_in[2];
    const float* n_prev = (const float*)d_in[3];
    const float* m_prev = (const float*)d_in[4];
    const float* Wi = (const float*)d_in[5];
    const float* bi = (const float*)d_in[6];
    const float* Wf = (const float*)d_in[7];
    const float* bf = (const float*)d_in[8];
    const float* Wo = (const float*)d_in[9];
    const float* bo = (const float*)d_in[10];
    const float* Wz = (const float*)d_in[11];
    const float* bz = (const float*)d_in[12];
    const float* gamma = (const float*)d_in[13];
    const float* beta  = (const float*)d_in[14];
    float* out = (float*)d_out;

    bf16_t* comb = (bf16_t*)d_ws;
    const size_t combBytes = (size_t)B_SZ * D_SZ * sizeof(bf16_t);  // ~25 MB
    bf16_t* W4 = (bf16_t*)((char*)d_ws + combBytes);                // ~50 MB

    pack_combined<<<dim3(D_SZ / 1024, B_SZ), 256, 0, stream>>>(x, h_prev, comb);
    pack_weights<<<dim3((unsigned)(((size_t)H_SZ * D_SZ) / 1024), G_SZ), 256, 0, stream>>>(
        Wi, Wf, Wo, Wz, W4);
    slstm_gemm<<<dim3(H_SZ / BN, B_SZ / BM), 256, 0, stream>>>(
        comb, W4, bi, bf, bo, bz, c_prev, n_prev, m_prev, out);
    layernorm_kernel<<<B_SZ, 256, 0, stream>>>(out, gamma, beta);
}
